// T5SelfAttention_12970801234307
// MI455X (gfx1250) — compile-verified, matrix operand order checked by reading, not run on hardware
//
#include <hip/hip_runtime.h>
#include <cstdint>
#include <cstddef>

// ---------------------------------------------------------------------------
// CDNA5 / gfx1250: wave32, WMMA 16x16x32 bf16 (f32 accum).
// ---------------------------------------------------------------------------
typedef __bf16 bf16;
typedef __attribute__((ext_vector_type(16))) __bf16 v16bf;
typedef __attribute__((ext_vector_type(8)))  __bf16 bf16x8;
typedef __attribute__((ext_vector_type(8)))  float  f32x8;
typedef int v4i_g __attribute__((vector_size(16)));   // matches builtin prototype

union AFrag { v16bf v; bf16x8 h[2]; };
union FAcc  { f32x8 v; float  f[8]; };

__device__ __forceinline__ f32x8 wmma_bf16(const AFrag& a, const AFrag& b, f32x8 c) {
  return __builtin_amdgcn_wmma_f32_16x16x32_bf16(false, a.v, false, b.v, (short)0, c,
                                                 false, false);
}

#define NEG_BIG (-3.0e38f)

// ---------------------------------------------------------------------------
// CDNA5 async memory->LDS staging (ASYNCcnt path). Guarded: if this toolchain
// doesn't declare the builtins, fall back to load-to-VGPR + ds_store.
// Per-lane 16B: global VGPR address -> LDS VGPR address. The builtin takes
// (v4i* global_src, v4i* lds_dst, imm offset, imm cpol); clang adapts pointer
// address spaces of builtin params to the argument's AS, so generic pointers
// are passed directly.
// ---------------------------------------------------------------------------
#if defined(__has_builtin)
#if __has_builtin(__builtin_amdgcn_global_load_async_to_lds_b128) && \
    __has_builtin(__builtin_amdgcn_s_wait_asynccnt)
#define USE_ASYNC_LDS 1
#endif
#endif

__device__ __forceinline__ void stage16(const bf16* g, bf16* l) {
#ifdef USE_ASYNC_LDS
  __builtin_amdgcn_global_load_async_to_lds_b128((v4i_g*)g, (v4i_g*)l, 0, 0);
#else
  *(bf16x8*)l = *(const bf16x8*)g;
#endif
}

__device__ __forceinline__ void stage_wait() {
#ifdef USE_ASYNC_LDS
  __builtin_amdgcn_s_wait_asynccnt(0);
#endif
}

// ---------------------------------------------------------------------------
// f32 -> bf16 elementwise convert (weights, one-time)
// ---------------------------------------------------------------------------
__global__ void k_f32_to_bf16(const float* __restrict__ in, bf16* __restrict__ out,
                              long long n) {
  long long i = (long long)blockIdx.x * blockDim.x + threadIdx.x;
  long long stride = (long long)gridDim.x * blockDim.x;
  for (; i < n; i += stride) out[i] = (bf16)in[i];
}

// ---------------------------------------------------------------------------
// T5 RMSNorm: one block per row; f32 stats, bf16 output. 256 threads = 8 waves.
// ---------------------------------------------------------------------------
__global__ void k_rmsnorm_bf16(const float* __restrict__ x, const float* __restrict__ w,
                               bf16* __restrict__ out, int D) {
  __shared__ float red[8];
  const int row = blockIdx.x;
  const float* xr = x + (size_t)row * D;
  float ss = 0.0f;
  for (int i = threadIdx.x; i < D; i += blockDim.x) { float v = xr[i]; ss += v * v; }
  for (int d = 1; d < 32; d <<= 1) ss += __shfl_xor(ss, d, 32);
  const int wave = threadIdx.x >> 5;
  if ((threadIdx.x & 31) == 0) red[wave] = ss;
  __syncthreads();
  float tot = 0.0f;
  for (int i = 0; i < 8; ++i) tot += red[i];
  const float rs = rsqrtf(tot / (float)D + 1e-6f);
  bf16* orow = out + (size_t)row * D;
  for (int i = threadIdx.x; i < D; i += blockDim.x) orow[i] = (bf16)(xr[i] * rs * w[i]);
}

// ---------------------------------------------------------------------------
// Tiled NT bf16 GEMM: C[M,N] = A[M,K] @ B[N,K]^T (+ optional f32 residual).
// Block 256 thr (8 waves); tile 128x128; wave tile 32x64 (2x4 WMMA 16x16).
// LDS rows padded to 40 bf16 (80B) so 16B fragment chunks stay aligned and
// bank-spread. K-tile = 32 (one WMMA K-step). Tile staging uses the CDNA5
// async memory->LDS path when available.
// ---------------------------------------------------------------------------
#define LDSA 40
__global__ void k_gemm_bf16_nt(const bf16* __restrict__ A, const bf16* __restrict__ B,
                               int M, int N, int K,
                               const float* __restrict__ resid,
                               float* __restrict__ Cf, bf16* __restrict__ Cb) {
  __shared__ __align__(16) bf16 sA[128 * LDSA];
  __shared__ __align__(16) bf16 sB[128 * LDSA];

  const int tid  = threadIdx.x;
  const int wave = tid >> 5, lane = tid & 31;
  const int wm = wave & 3, wn = wave >> 2;          // 4x2 wave grid
  const int l16 = lane & 15, hi = lane >> 4;        // ISA 16-bit A layout halves
  const int kh = hi * 8;                            // K base: 0 or 8 (then +16)
  const int bm = blockIdx.x * 128, bn = blockIdx.y * 128;

  const int ldrow = tid >> 1;                        // 0..127
  const int ldcol = (tid & 1) * 16;                  // 0 or 16

  FAcc acc[2][4];
  for (int m = 0; m < 2; ++m)
    for (int n = 0; n < 4; ++n)
      for (int i = 0; i < 8; ++i) acc[m][n].f[i] = 0.0f;

  for (int k0 = 0; k0 < K; k0 += 32) {
    __syncthreads();
    const bf16* ga = A + (size_t)(bm + ldrow) * K + k0 + ldcol;
    const bf16* gb = B + (size_t)(bn + ldrow) * K + k0 + ldcol;
    stage16(ga,     &sA[ldrow * LDSA + ldcol]);
    stage16(ga + 8, &sA[ldrow * LDSA + ldcol + 8]);
    stage16(gb,     &sB[ldrow * LDSA + ldcol]);
    stage16(gb + 8, &sB[ldrow * LDSA + ldcol + 8]);
    stage_wait();
    __syncthreads();

    AFrag a[2], b[4];
    for (int ms = 0; ms < 2; ++ms) {
      const int r = wm * 32 + ms * 16 + l16;
      a[ms].h[0] = *(const bf16x8*)&sA[r * LDSA + kh];
      a[ms].h[1] = *(const bf16x8*)&sA[r * LDSA + kh + 16];
    }
    for (int ns = 0; ns < 4; ++ns) {
      const int r = wn * 64 + ns * 16 + l16;
      b[ns].h[0] = *(const bf16x8*)&sB[r * LDSA + kh];
      b[ns].h[1] = *(const bf16x8*)&sB[r * LDSA + kh + 16];
    }
    for (int ms = 0; ms < 2; ++ms)
      for (int ns = 0; ns < 4; ++ns)
        acc[ms][ns].v = wmma_bf16(a[ms], b[ns], acc[ms][ns].v);
  }

  // Epilogue (C/D layout: lane = column, VGPR r = row r + 8*hi)
  for (int ms = 0; ms < 2; ++ms)
    for (int ns = 0; ns < 4; ++ns) {
      const int row0 = bm + wm * 32 + ms * 16 + hi * 8;
      const int col  = bn + wn * 64 + ns * 16 + l16;
      for (int r = 0; r < 8; ++r) {
        const size_t idx = (size_t)(row0 + r) * N + col;
        float v = acc[ms][ns].f[r];
        if (resid) v += resid[idx];
        if (Cf) Cf[idx] = v;
        if (Cb) Cb[idx] = (bf16)v;
      }
    }
}

// ---------------------------------------------------------------------------
// Flash attention (T5: no 1/sqrt(c), +pos_bias, mask). c = 64.
// Block = 128 thr (4 waves); each wave owns 16 query rows; block = 64 q-rows
// of one head. Streams K/V in 32-key LDS tiles; online softmax; P routed via
// per-wave LDS scratch into WMMA A-layout for P@V.
// ---------------------------------------------------------------------------
#define KLDS 72
#define VLDS 40
#define PLDS 40
__global__ void k_flash_attn(const bf16* __restrict__ Q, const bf16* __restrict__ Km,
                             const bf16* __restrict__ Vm,
                             const float* __restrict__ bias,
                             const int* __restrict__ mask,
                             bf16* __restrict__ AO, int L, int D) {
  __shared__ __align__(16) bf16 sK[32 * KLDS];
  __shared__ __align__(16) bf16 sVt[64 * VLDS];      // transposed: [c][key]
  __shared__ __align__(16) bf16 sP[4 * 16 * PLDS];   // per-wave P scratch

  const int tid  = threadIdx.x;
  const int wave = tid >> 5, lane = tid & 31;
  const int l16 = lane & 15, hi = lane >> 4;
  const int kh = hi * 8;
  const int h = blockIdx.y;
  const int qbase = blockIdx.x * 64 + wave * 16;

  // Resident Q fragments (16 x 64 bf16 -> two K=32 A-frags)
  AFrag qa[2];
  {
    const bf16* qp = Q + (size_t)(qbase + l16) * D + h * 64;
    for (int f = 0; f < 2; ++f) {
      qa[f].h[0] = *(const bf16x8*)(qp + f * 32 + kh);
      qa[f].h[1] = *(const bf16x8*)(qp + f * 32 + kh + 16);
    }
  }

  FAcc o[4];
  for (int n = 0; n < 4; ++n)
    for (int i = 0; i < 8; ++i) o[n].f[i] = 0.0f;
  float mrow[8], lrow[8];
  for (int r = 0; r < 8; ++r) { mrow[r] = NEG_BIG; lrow[r] = 0.0f; }

  const int krow = tid >> 2;            // 0..31
  const int kcb  = (tid & 3) * 16;      // 0,16,32,48
  bf16* pw = &sP[wave * 16 * PLDS];

  for (int j0 = 0; j0 < L; j0 += 32) {
    __syncthreads();
    // ---- stage K tile (row-major, async->LDS path) ----
    const bf16* kp = Km + (size_t)(j0 + krow) * D + h * 64 + kcb;
    stage16(kp,     &sK[krow * KLDS + kcb]);
    stage16(kp + 8, &sK[krow * KLDS + kcb + 8]);
    // ---- stage V tile transposed (per-element scatter, VALU path) ----
    const bf16* vp = Vm + (size_t)(j0 + krow) * D + h * 64 + kcb;
    union { bf16x8 v; bf16 e[8]; } t0, t1;
    t0.v = *(const bf16x8*)vp;
    t1.v = *(const bf16x8*)(vp + 8);
    for (int i = 0; i < 8; ++i) {
      sVt[(kcb + i) * VLDS + krow]     = t0.e[i];
      sVt[(kcb + 8 + i) * VLDS + krow] = t1.e[i];
    }
    // Prefetch next pos_bias tile through the gfx1250 prefetch path.
    if (j0 + 32 < L)
      __builtin_prefetch(bias + ((size_t)h * L + qbase) * L + j0 + 32, 0, 1);
    stage_wait();
    __syncthreads();

    // ---- S = Q @ K^T  (16 x 32 per wave) ----
    FAcc s[2];
    for (int ns = 0; ns < 2; ++ns) {
      for (int i = 0; i < 8; ++i) s[ns].f[i] = 0.0f;
      const int rn = ns * 16 + l16;
      AFrag b0, b1;
      b0.h[0] = *(const bf16x8*)&sK[rn * KLDS + kh];
      b0.h[1] = *(const bf16x8*)&sK[rn * KLDS + kh + 16];
      b1.h[0] = *(const bf16x8*)&sK[rn * KLDS + 32 + kh];
      b1.h[1] = *(const bf16x8*)&sK[rn * KLDS + 32 + kh + 16];
      s[ns].v = wmma_bf16(qa[0], b0, s[ns].v);
      s[ns].v = wmma_bf16(qa[1], b1, s[ns].v);
    }

    // ---- + pos_bias, mask ----
    for (int ns = 0; ns < 2; ++ns) {
      const int j = j0 + ns * 16 + l16;
      const bool mz = (mask[j] == 0);
      const float* bp = bias + ((size_t)h * L + (qbase + hi * 8)) * L + j;
      for (int r = 0; r < 8; ++r) {
        if (mz) s[ns].f[r] = NEG_BIG;
        else    s[ns].f[r] += bp[(size_t)r * L];
      }
    }

    // ---- online softmax (row stats per accumulator register; reductions stay
    //      inside each 16-lane group: xor masks 1,2,4,8 — wave32-correct) ----
    for (int r = 0; r < 8; ++r) {
      float mx = fmaxf(s[0].f[r], s[1].f[r]);
      for (int d = 1; d < 16; d <<= 1) mx = fmaxf(mx, __shfl_xor(mx, d, 32));
      const float mnew = fmaxf(mrow[r], mx);
      const float corr = __expf(mrow[r] - mnew);
      mrow[r] = mnew;
      const float p0 = __expf(s[0].f[r] - mnew);
      const float p1 = __expf(s[1].f[r] - mnew);
      s[0].f[r] = p0; s[1].f[r] = p1;
      float ps = p0 + p1;
      for (int d = 1; d < 16; d <<= 1) ps += __shfl_xor(ps, d, 32);
      lrow[r] = lrow[r] * corr + ps;
      for (int n = 0; n < 4; ++n) o[n].f[r] *= corr;
    }

    // ---- P (C-layout) -> LDS -> A-layout fragment ----
    for (int ns = 0; ns < 2; ++ns) {
      const int col = ns * 16 + l16;
      for (int r = 0; r < 8; ++r) pw[(r + hi * 8) * PLDS + col] = (bf16)s[ns].f[r];
    }
    AFrag pa;
    pa.h[0] = *(const bf16x8*)&pw[l16 * PLDS + kh];
    pa.h[1] = *(const bf16x8*)&pw[l16 * PLDS + kh + 16];

    // ---- O += P @ V  (B operand from transposed V tile) ----
    for (int ns = 0; ns < 4; ++ns) {
      const int rn = ns * 16 + l16;
      AFrag vb;
      vb.h[0] = *(const bf16x8*)&sVt[rn * VLDS + kh];
      vb.h[1] = *(const bf16x8*)&sVt[rn * VLDS + kh + 16];
      o[ns].v = wmma_bf16(pa, vb, o[ns].v);
    }
  }

  // ---- finalize: O / l, write bf16 into [L, H*c] for the Wo GEMM ----
  for (int ns = 0; ns < 4; ++ns)
    for (int r = 0; r < 8; ++r) {
      const int row = qbase + hi * 8 + r;
      const int col = h * 64 + ns * 16 + l16;
      AO[(size_t)row * D + col] = (bf16)(o[ns].f[r] / lrow[r]);
    }
}

// ---------------------------------------------------------------------------
// Dual GEMM FFN: h = (A@FC1^T) * gelu_tanh(A@GATE^T), bf16 out.
// Block 256 thr; tile 128(M) x 64(N); wave tile 32x32; shared A traffic.
// ---------------------------------------------------------------------------
__global__ void k_ffn_dual_gemm(const bf16* __restrict__ A, const bf16* __restrict__ B1,
                                const bf16* __restrict__ B2, int M, int N, int K,
                                bf16* __restrict__ Hout) {
  __shared__ __align__(16) bf16 sA[128 * LDSA];
  __shared__ __align__(16) bf16 sB1[64 * LDSA];
  __shared__ __align__(16) bf16 sB2[64 * LDSA];

  const int tid  = threadIdx.x;
  const int wave = tid >> 5, lane = tid & 31;
  const int wm = wave & 3, wn = wave >> 2;          // 4x2
  const int l16 = lane & 15, hi = lane >> 4;
  const int kh = hi * 8;
  const int bm = blockIdx.x * 128, bn = blockIdx.y * 64;

  const int arow = tid >> 1, acol = (tid & 1) * 16;
  const int brow = tid >> 2, bcol = (tid & 3) * 8;

  FAcc a1[2][2], a2[2][2];
  for (int m = 0; m < 2; ++m)
    for (int n = 0; n < 2; ++n)
      for (int i = 0; i < 8; ++i) { a1[m][n].f[i] = 0.0f; a2[m][n].f[i] = 0.0f; }

  for (int k0 = 0; k0 < K; k0 += 32) {
    __syncthreads();
    const bf16* ga = A + (size_t)(bm + arow) * K + k0 + acol;
    stage16(ga,     &sA[arow * LDSA + acol]);
    stage16(ga + 8, &sA[arow * LDSA + acol + 8]);
    *(bf16x8*)&sB1[brow * LDSA + bcol] =
        *(const bf16x8*)(B1 + (size_t)(bn + brow) * K + k0 + bcol);
    *(bf16x8*)&sB2[brow * LDSA + bcol] =
        *(const bf16x8*)(B2 + (size_t)(bn + brow) * K + k0 + bcol);
    stage_wait();
    __syncthreads();

    AFrag af[2], b1f[2], b2f[2];
    for (int ms = 0; ms < 2; ++ms) {
      const int r = wm * 32 + ms * 16 + l16;
      af[ms].h[0] = *(const bf16x8*)&sA[r * LDSA + kh];
      af[ms].h[1] = *(const bf16x8*)&sA[r * LDSA + kh + 16];
    }
    for (int ns = 0; ns < 2; ++ns) {
      const int r = wn * 32 + ns * 16 + l16;
      b1f[ns].h[0] = *(const bf16x8*)&sB1[r * LDSA + kh];
      b1f[ns].h[1] = *(const bf16x8*)&sB1[r * LDSA + kh + 16];
      b2f[ns].h[0] = *(const bf16x8*)&sB2[r * LDSA + kh];
      b2f[ns].h[1] = *(const bf16x8*)&sB2[r * LDSA + kh + 16];
    }
    for (int ms = 0; ms < 2; ++ms)
      for (int ns = 0; ns < 2; ++ns) {
        a1[ms][ns].v = wmma_bf16(af[ms], b1f[ns], a1[ms][ns].v);
        a2[ms][ns].v = wmma_bf16(af[ms], b2f[ns], a2[ms][ns].v);
      }
  }

  for (int ms = 0; ms < 2; ++ms)
    for (int ns = 0; ns < 2; ++ns) {
      const int row0 = bm + wm * 32 + ms * 16 + hi * 8;
      const int col  = bn + wn * 32 + ns * 16 + l16;
      for (int r = 0; r < 8; ++r) {
        const float v1 = a1[ms][ns].f[r];
        const float v2 = a2[ms][ns].f[r];
        const float g  = 0.5f * v2 *
            (1.0f + tanhf(0.7978845608028654f * (v2 + 0.044715f * v2 * v2 * v2)));
        Hout[(size_t)(row0 + r) * N + col] = (bf16)(v1 * g);
      }
    }
}

// ---------------------------------------------------------------------------
// Host orchestration
// ---------------------------------------------------------------------------
extern "C" void kernel_launch(void* const* d_in, const int* in_sizes, int n_in,
                              void* d_out, int out_size, void* d_ws, size_t ws_size,
                              hipStream_t stream) {
  const float* x      = (const float*)d_in[0];
  const int*   mask   = (const int*)  d_in[1];
  const float* bias   = (const float*)d_in[2];
  const float* n1w    = (const float*)d_in[3];
  const float* wq     = (const float*)d_in[4];
  const float* wk     = (const float*)d_in[5];
  const float* wv     = (const float*)d_in[6];
  const float* wo     = (const float*)d_in[7];
  const float* n2w    = (const float*)d_in[8];
  const float* fc1    = (const float*)d_in[9];
  const float* gate   = (const float*)d_in[10];
  const float* fc2    = (const float*)d_in[11];

  const int D    = in_sizes[3];
  const int L    = in_sizes[0] / D;
  const int H    = (int)((long long)in_sizes[2] / ((long long)L * L));
  const int Dffn = in_sizes[9] / D;

  // Workspace carve-up (256B aligned)
  char* base = (char*)d_ws;
  size_t off = 0;
  auto carve = [&](size_t bytes) -> char* {
    char* p = base + off;
    off += (bytes + 255) & ~(size_t)255;
    return p;
  };
  const size_t LD = (size_t)L * D, DD = (size_t)D * D, DF = (size_t)Dffn * D;
  bf16* xn_b   = (bf16*)carve(LD * 2);
  bf16* wq_b   = (bf16*)carve(DD * 2);
  bf16* wk_b   = (bf16*)carve(DD * 2);
  bf16* wv_b   = (bf16*)carve(DD * 2);
  bf16* wo_b   = (bf16*)carve(DD * 2);
  bf16* fc1_b  = (bf16*)carve(DF * 2);
  bf16* gate_b = (bf16*)carve(DF * 2);
  bf16* fc2_b  = (bf16*)carve(DF * 2);
  bf16* q_b    = (bf16*)carve(LD * 2);
  bf16* k_b    = (bf16*)carve(LD * 2);
  bf16* v_b    = (bf16*)carve(LD * 2);
  bf16* ao_b   = (bf16*)carve(LD * 2);
  float* x1    = (float*)carve(LD * 4);
  bf16* xn2_b  = (bf16*)carve(LD * 2);
  bf16* h_b    = (bf16*)carve((size_t)L * Dffn * 2);
  (void)ws_size; (void)n_in; (void)out_size;

  const dim3 cvt_grid(2048), cvt_blk(256);
  k_f32_to_bf16<<<cvt_grid, cvt_blk, 0, stream>>>(wq,   wq_b,   (long long)DD);
  k_f32_to_bf16<<<cvt_grid, cvt_blk, 0, stream>>>(wk,   wk_b,   (long long)DD);
  k_f32_to_bf16<<<cvt_grid, cvt_blk, 0, stream>>>(wv,   wv_b,   (long long)DD);
  k_f32_to_bf16<<<cvt_grid, cvt_blk, 0, stream>>>(wo,   wo_b,   (long long)DD);
  k_f32_to_bf16<<<cvt_grid, cvt_blk, 0, stream>>>(fc1,  fc1_b,  (long long)DF);
  k_f32_to_bf16<<<cvt_grid, cvt_blk, 0, stream>>>(gate, gate_b, (long long)DF);
  k_f32_to_bf16<<<cvt_grid, cvt_blk, 0, stream>>>(fc2,  fc2_b,  (long long)DF);

  // norm1
  k_rmsnorm_bf16<<<dim3(L), dim3(256), 0, stream>>>(x, n1w, xn_b, D);

  // Q/K/V projections
  const dim3 gqkv(L / 128, D / 128), gblk(256);
  k_gemm_bf16_nt<<<gqkv, gblk, 0, stream>>>(xn_b, wq_b, L, D, D, nullptr, nullptr, q_b);
  k_gemm_bf16_nt<<<gqkv, gblk, 0, stream>>>(xn_b, wk_b, L, D, D, nullptr, nullptr, k_b);
  k_gemm_bf16_nt<<<gqkv, gblk, 0, stream>>>(xn_b, wv_b, L, D, D, nullptr, nullptr, v_b);

  // flash attention
  k_flash_attn<<<dim3(L / 64, H), dim3(128), 0, stream>>>(q_b, k_b, v_b, bias, mask,
                                                          ao_b, L, D);

  // x1 = x + ao @ wo^T
  k_gemm_bf16_nt<<<gqkv, gblk, 0, stream>>>(ao_b, wo_b, L, D, D, x, x1, nullptr);

  // norm2
  k_rmsnorm_bf16<<<dim3(L), dim3(256), 0, stream>>>(x1, n2w, xn2_b, D);

  // h = (xn2 @ fc1^T) * gelu(xn2 @ gate^T)
  k_ffn_dual_gemm<<<dim3(L / 128, Dffn / 64), gblk, 0, stream>>>(xn2_b, fc1_b, gate_b,
                                                                 L, Dffn, D, h_b);

  // out = x1 + h @ fc2^T
  k_gemm_bf16_nt<<<gqkv, gblk, 0, stream>>>(h_b, fc2_b, L, D, Dffn, x1,
                                            (float*)d_out, nullptr);
}
